// MultiHeadedLatentAttention_74217034875658
// MI455X (gfx1250) — matrix operand order repaired and checked by reading, not compiled, not run on hardware
//
#include <hip/hip_runtime.h>
#include <hip/hip_bf16.h>

// ---------------------------------------------------------------------------
// MLA forward for MI455X (gfx1250, wave32, WMMA).
//  * all GEMM operands pre-converted to bf16 once (halves steady-state HBM
//    traffic; strips v_cvt VALU out of the GEMM hot loop)
//  * v_wmma_f32_16x16x32_bf16 everywhere, f32 accumulation, f32 softmax
//  * flash-attention streaming (no S x S tensor): K/V (~42MB bf16) fits in
//    the 192MB L2, so repeated K-sweeps hit L2, not HBM
//  * A-tiles staged global->LDS with GLOBAL_LOAD_ASYNC_TO_LDS_B128
//    (ASYNCcnt-tracked, no VGPR round trip)
// ---------------------------------------------------------------------------

typedef __attribute__((ext_vector_type(16))) __bf16 v16bf;
typedef __attribute__((ext_vector_type(8)))  float  v8f;
typedef int v4i __attribute__((ext_vector_type(4)));

constexpr int kB   = 2;
constexpr int kS   = 2048;
constexpr int kH   = 16;
constexpr int kHD  = 128;
constexpr int kRD  = 64;
constexpr int kQKD = 192;                 // HD + RD
constexpr int kRows = kB * kS;            // 4096

#if __has_builtin(__builtin_amdgcn_global_load_async_to_lds_b128) && \
    __has_builtin(__builtin_amdgcn_s_wait_asynccnt)
#define USE_ASYNC_LDS 1
#else
#define USE_ASYNC_LDS 0
#endif

__device__ inline v8f wmma_bf16(const v16bf& a, const v16bf& b, const v8f& c) {
  return __builtin_amdgcn_wmma_f32_16x16x32_bf16(false, a, false, b,
                                                 (short)0, c, false, false);
}

// ---------------------------------------------------------------------------
// f32 -> bf16 one-time conversion
// ---------------------------------------------------------------------------
__global__ __launch_bounds__(256)
void to_bf16(const float* __restrict__ src, __bf16* __restrict__ dst,
             long long n) {
  const long long idx = (long long)blockIdx.x * blockDim.x + threadIdx.x;
  if (idx < n) dst[idx] = (__bf16)src[idx];
}

// ---------------------------------------------------------------------------
// GEMM: C[M,N] = A[M,K] * B[K,N] (+f32 bias), A/B bf16, OutT in {float,bf16}.
// Block tile 128x128, BK=32, 256 threads = 8 waves, wave tile 64x32.
// All M % 128 == 0, N % 128 == 0, K % 32 == 0 below.
// ---------------------------------------------------------------------------
#define BM 128
#define BN 128
#define BK 32
#define LDP 40   // padded LDS row stride (bf16), keeps 16B alignment

template <typename OutT>
__global__ __launch_bounds__(256)
void gemm_bf16(const __bf16* __restrict__ A, const __bf16* __restrict__ Bm,
               OutT* __restrict__ C, const float* __restrict__ bias,
               int M, int N, int K) {
  __shared__ __bf16 As[BM][LDP];   // [m][k]
  __shared__ __bf16 Bs[BN][LDP];   // transposed: [n][k] -> contiguous along K
  const int tid  = threadIdx.x;
  const int lane = tid & 31;
  const int wave = tid >> 5;
  const int bm = blockIdx.y * BM;
  const int bn = blockIdx.x * BN;
  const int wm = (wave >> 2) * 64;
  const int wn = (wave & 3) * 32;

  const v8f zero8 = {0.f, 0.f, 0.f, 0.f, 0.f, 0.f, 0.f, 0.f};
  v8f acc[4][2];
#pragma unroll
  for (int mi = 0; mi < 4; ++mi)
#pragma unroll
    for (int ni = 0; ni < 2; ++ni) acc[mi][ni] = zero8;

  const int brow = tid >> 3;            // B copy: k 0..31
  const int bcol = (tid & 7) * 16;      // B copy: n 0..112

  const int laneM = lane & 15;
  const int ko8   = (lane >> 4) * 8;    // unused; A-fragment also reads at kh16
  const int kh16  = (lane >> 4) * 16;   // B-fragment K phase

  for (int k0 = 0; k0 < K; k0 += BK) {
    // ---- A tile 128x32 bf16 = 512 x 16B chunks, 2 per thread ----
#pragma unroll
    for (int i = 0; i < 2; ++i) {
      const int c   = tid + i * 256;
      const int row = c >> 2;           // 4 x 16B chunks per 64B row
      const int col = (c & 3) * 8;
      const __bf16* gp = A + (size_t)(bm + row) * K + k0 + col;
#if USE_ASYNC_LDS
      __builtin_amdgcn_global_load_async_to_lds_b128(
          (__attribute__((address_space(1))) v4i*)gp,
          (__attribute__((address_space(3))) v4i*)&As[row][col], 0, 0);
#else
      *(v4i*)&As[row][col] = *(const v4i*)gp;
#endif
    }
    // ---- B tile 32x128 bf16, transposed into LDS ----
    {
      const __bf16* gp = Bm + (size_t)(k0 + brow) * N + bn + bcol;
      __bf16 t[16];
      *(v4i*)&t[0] = *(const v4i*)gp;
      *(v4i*)&t[8] = *(const v4i*)(gp + 8);
#pragma unroll
      for (int j = 0; j < 16; ++j) Bs[bcol + j][brow] = t[j];
    }
    // ---- prefetch next K-tile into L2 (global_prefetch_b8) ----
    if (k0 + BK < K) {
      __builtin_prefetch(A + (size_t)(bm + (tid >> 2)) * K + k0 + BK, 0, 3);
      __builtin_prefetch(Bm + (size_t)(k0 + BK + brow) * N + bn + bcol, 0, 3);
    }
#if USE_ASYNC_LDS
    __builtin_amdgcn_s_wait_asynccnt(0);
#endif
    __syncthreads();

    v16bf afr[4], bfr[2];
#pragma unroll
    for (int mi = 0; mi < 4; ++mi) {
      const int row = wm + mi * 16 + laneM;
#pragma unroll
      for (int j = 0; j < 16; ++j) afr[mi][j] = As[row][kh16 + j];
    }
#pragma unroll
    for (int ni = 0; ni < 2; ++ni) {
      const int col = wn + ni * 16 + laneM;
#pragma unroll
      for (int j = 0; j < 16; ++j) bfr[ni][j] = Bs[col][kh16 + j];
    }
#pragma unroll
    for (int mi = 0; mi < 4; ++mi)
#pragma unroll
      for (int ni = 0; ni < 2; ++ni)
        acc[mi][ni] = wmma_bf16(afr[mi], bfr[ni], acc[mi][ni]);
    __syncthreads();
  }

  // epilogue (C layout: VGPR r -> row r for lanes 0-15, row 8+r for 16-31)
#pragma unroll
  for (int mi = 0; mi < 4; ++mi)
#pragma unroll
    for (int ni = 0; ni < 2; ++ni) {
      const int col = bn + wn + ni * 16 + laneM;
      const float bv = bias ? bias[col] : 0.f;
#pragma unroll
      for (int r = 0; r < 8; ++r) {
        const int row = bm + wm + mi * 16 + r + ((lane >> 4) << 3);
        C[(size_t)row * N + col] = (OutT)(acc[mi][ni][r] + bv);
      }
    }
}

// ---------------------------------------------------------------------------
// RoPE + concat: bf16 [rows][H*192] from nope bf16 [rows][H*128] and rope
// part bf16 [rows][H*64] (interleaved re/im pairs), cos/sin f32 [S][32].
// ---------------------------------------------------------------------------
__global__ __launch_bounds__(256)
void rope_concat(const __bf16* __restrict__ qh, const __bf16* __restrict__ qr,
                 const float* __restrict__ cosb, const float* __restrict__ sinb,
                 __bf16* __restrict__ outb) {
  const long long idx = (long long)blockIdx.x * blockDim.x + threadIdx.x;
  const long long total = (long long)kRows * kH * kQKD;
  if (idx >= total) return;
  const int d = (int)(idx % kQKD);
  const long long t = idx / kQKD;
  const int h = (int)(t % kH);
  const long long row = t / kH;
  float val;
  if (d < kHD) {
    val = (float)qh[row * (kH * kHD) + h * kHD + d];
  } else {
    const int d2 = d - kHD;
    const int p  = d2 >> 1;
    const int s  = (int)(row % kS);
    const float c  = cosb[s * (kRD / 2) + p];
    const float sn = sinb[s * (kRD / 2) + p];
    const float re = (float)qr[row * (kH * kRD) + h * kRD + 2 * p];
    const float im = (float)qr[row * (kH * kRD) + h * kRD + 2 * p + 1];
    val = (d2 & 1) ? (re * sn + im * c) : (re * c - im * sn);
  }
  outb[idx] = (__bf16)val;
}

// ---------------------------------------------------------------------------
// V: bf16 [b*s][h*128+d] -> bf16 [b][h][d][s] (tokens contiguous, so PV
// B-fragments load 16 contiguous bf16 per lane).
// ---------------------------------------------------------------------------
__global__ __launch_bounds__(256)
void v_transpose(const __bf16* __restrict__ vsrc, __bf16* __restrict__ vt) {
  const long long idx = (long long)blockIdx.x * blockDim.x + threadIdx.x;
  const long long total = (long long)kRows * kH * kHD;
  if (idx >= total) return;
  const int c = (int)(idx % (kH * kHD));
  const long long row = idx / (kH * kHD);
  const int h = c / kHD, d = c % kHD;
  const long long b = row / kS, s = row % kS;
  vt[(((b * kH + h) * kHD) + d) * kS + s] = vsrc[idx];
}

// ---------------------------------------------------------------------------
// Flash attention. One wave owns 16 query rows; block = 8 waves = 128 rows.
// 32-token causal K tiles: scores via 6x2 WMMA (K=192), f32 online softmax
// (16-lane shfl_xor row reductions per the C-fragment layout), P re-laid out
// C-frag -> A-frag through a per-wave LDS patch, 8 WMMA for P*V.
// grid = (S/128, B*H). Output bf16 (feeds the Wo GEMM).
// ---------------------------------------------------------------------------
__device__ inline float rmax16(float v) {
  v = fmaxf(v, __shfl_xor(v, 1));
  v = fmaxf(v, __shfl_xor(v, 2));
  v = fmaxf(v, __shfl_xor(v, 4));
  v = fmaxf(v, __shfl_xor(v, 8));
  return v;
}
__device__ inline float rsum16(float v) {
  v += __shfl_xor(v, 1);
  v += __shfl_xor(v, 2);
  v += __shfl_xor(v, 4);
  v += __shfl_xor(v, 8);
  return v;
}

__global__ __launch_bounds__(256)
void flash_attn(const __bf16* __restrict__ qf, const __bf16* __restrict__ kf,
                const __bf16* __restrict__ vt, __bf16* __restrict__ out) {
  __shared__ __bf16 Plds[8][16][32];    // per-wave private P patch
  const int tid  = threadIdx.x;
  const int lane = tid & 31;
  const int wave = tid >> 5;
  const int bh = blockIdx.y;
  const int b  = bh >> 4;
  const int h  = bh & 15;
  const int q0 = blockIdx.x * 128 + wave * 16;
  const size_t qrowbase = (size_t)b * kS + q0;
  const int hoff = h * kQKD;
  const int laneM = lane & 15;
  const int ko8   = (lane >> 4) * 8;
  const int kh16  = (lane >> 4) * 16;
  const float scale = 0.07216878364870323f;      // 1/sqrt(192)
  const float NEG = -3.0e30f;

  v16bf qfr[6];
  {
    const __bf16* qp = qf + (qrowbase + laneM) * (kH * kQKD) + hoff;
#pragma unroll
    for (int kk = 0; kk < 6; ++kk)
#pragma unroll
      for (int j = 0; j < 16; ++j) qfr[kk][j] = qp[kk * 32 + kh16 + j];
  }

  const v8f zero8 = {0.f, 0.f, 0.f, 0.f, 0.f, 0.f, 0.f, 0.f};
  v8f o[8];
#pragma unroll
  for (int n = 0; n < 8; ++n) o[n] = zero8;
  float mrow[8], lrow[8];
#pragma unroll
  for (int r = 0; r < 8; ++r) { mrow[r] = NEG; lrow[r] = 0.f; }

  for (int kb = 0; kb < q0 + 16; kb += 32) {
    v8f sc[2];
#pragma unroll
    for (int nt = 0; nt < 2; ++nt) {
      v8f s = zero8;
      const int tok = kb + nt * 16 + laneM;
      const __bf16* kp = kf + ((size_t)b * kS + tok) * (kH * kQKD) + hoff + kh16;
#pragma unroll
      for (int kk = 0; kk < 6; ++kk) {
        v16bf bfr;
#pragma unroll
        for (int j = 0; j < 16; ++j) bfr[j] = kp[kk * 32 + j];
        s = wmma_bf16(qfr[kk], bfr, s);
      }
      sc[nt] = s;
    }
#pragma unroll
    for (int r = 0; r < 8; ++r) {
      const int row = q0 + r + ((lane >> 4) << 3);
      float v0 = (kb + laneM <= row)      ? sc[0][r] * scale : NEG;
      float v1 = (kb + 16 + laneM <= row) ? sc[1][r] * scale : NEG;
      const float mnew = fmaxf(mrow[r], rmax16(fmaxf(v0, v1)));
      const float corr = __expf(mrow[r] - mnew);
      const float p0 = __expf(v0 - mnew);
      const float p1 = __expf(v1 - mnew);
      sc[0][r] = p0; sc[1][r] = p1;
      lrow[r] = lrow[r] * corr + rsum16(p0 + p1);
      mrow[r] = mnew;
#pragma unroll
      for (int n = 0; n < 8; ++n) o[n][r] *= corr;
    }
#pragma unroll
    for (int nt = 0; nt < 2; ++nt)
#pragma unroll
      for (int r = 0; r < 8; ++r)
        Plds[wave][r + ((lane >> 4) << 3)][nt * 16 + laneM] = (__bf16)sc[nt][r];
    v16bf pfr;
#pragma unroll
    for (int j = 0; j < 16; ++j) pfr[j] = Plds[wave][laneM][kh16 + j];
#pragma unroll
    for (int n = 0; n < 8; ++n) {
      const int d = n * 16 + laneM;
      const __bf16* vp = vt + (((size_t)bh * kHD) + d) * kS + kb + kh16;
      v16bf vfr;
#pragma unroll
      for (int j = 0; j < 16; ++j) vfr[j] = vp[j];
      o[n] = wmma_bf16(pfr, vfr, o[n]);
    }
  }

#pragma unroll
  for (int n = 0; n < 8; ++n) {
    const int d = n * 16 + laneM;
#pragma unroll
    for (int r = 0; r < 8; ++r) {
      const int row = r + ((lane >> 4) << 3);
      out[(qrowbase + row) * (kH * kHD) + h * kHD + d] =
          (__bf16)(o[n][r] / lrow[r]);
    }
  }
}

// ---------------------------------------------------------------------------
// Host orchestration. Workspace (bytes, ~160MB total):
//   xb    bf16 4096x2048 @ 0
//   Wlqb  bf16 2048x1536 @  16777216
//   Wkvb  bf16 2048x512  @  23068672
//   Wqb   bf16 1536x2048 @  25165824
//   Wqrb  bf16 1536x1024 @  31457280
//   Wkb   bf16  512x2048 @  34603008
//   Wvb   bf16  512x2048 @  36700160
//   Wkrb  bf16 2048x1024 @  38797312
//   Wob   bf16 2048x2048 @  42991616
//   cqb   bf16 4096x1536 @  51380224
//   ckvb  bf16 4096x512  @  63963136
//   tmpb  bf16 4096x2048 @  68157440   (reused: q, k, v, attn-out)
//   tmprb bf16 4096x1024 @  84934656   (reused: qr, kr)
//   qfb   bf16 4096x3072 @  93323264
//   kfb   bf16 4096x3072 @ 118489088
//   vtb   bf16 4096x2048 @ 143654912   (end 160432128)
// ---------------------------------------------------------------------------
extern "C" void kernel_launch(void* const* d_in, const int* in_sizes, int n_in,
                              void* d_out, int out_size, void* d_ws, size_t ws_size,
                              hipStream_t stream) {
  const float* x    = (const float*)d_in[0];
  const float* fcos = (const float*)d_in[1];
  const float* fsin = (const float*)d_in[2];
  const float* Wkv  = (const float*)d_in[3];
  const float* Wlq  = (const float*)d_in[4];
  const float* Wq   = (const float*)d_in[5];
  const float* Wk   = (const float*)d_in[6];
  const float* Wv   = (const float*)d_in[7];
  const float* Wqr  = (const float*)d_in[8];
  const float* bqr  = (const float*)d_in[9];
  const float* Wkr  = (const float*)d_in[10];
  const float* bkr  = (const float*)d_in[11];
  const float* Wo   = (const float*)d_in[12];
  const float* bo   = (const float*)d_in[13];
  float* outp = (float*)d_out;

  char* ws = (char*)d_ws;
  __bf16* xb    = (__bf16*)(ws + 0ull);
  __bf16* Wlqb  = (__bf16*)(ws + 16777216ull);
  __bf16* Wkvb  = (__bf16*)(ws + 23068672ull);
  __bf16* Wqb   = (__bf16*)(ws + 25165824ull);
  __bf16* Wqrb  = (__bf16*)(ws + 31457280ull);
  __bf16* Wkb   = (__bf16*)(ws + 34603008ull);
  __bf16* Wvb   = (__bf16*)(ws + 36700160ull);
  __bf16* Wkrb  = (__bf16*)(ws + 38797312ull);
  __bf16* Wob   = (__bf16*)(ws + 42991616ull);
  __bf16* cqb   = (__bf16*)(ws + 51380224ull);
  __bf16* ckvb  = (__bf16*)(ws + 63963136ull);
  __bf16* tmpb  = (__bf16*)(ws + 68157440ull);
  __bf16* tmprb = (__bf16*)(ws + 84934656ull);
  __bf16* qfb   = (__bf16*)(ws + 93323264ull);
  __bf16* kfb   = (__bf16*)(ws + 118489088ull);
  __bf16* vtb   = (__bf16*)(ws + 143654912ull);

  const dim3 blk(256);
  auto cvt = [&](const float* s, __bf16* d, long long n) {
    to_bf16<<<(unsigned)((n + 255) / 256), blk, 0, stream>>>(s, d, n);
  };
  auto gemm_b = [&](const __bf16* A, const __bf16* Bmat, __bf16* C,
                    const float* bias, int M, int N, int K) {
    dim3 grid(N / BN, M / BM);
    gemm_bf16<__bf16><<<grid, blk, 0, stream>>>(A, Bmat, C, bias, M, N, K);
  };

  // one-time bf16 conversion of x and all weights
  cvt(x,   xb,   (long long)kRows * 2048);
  cvt(Wlq, Wlqb, 2048LL * 1536);
  cvt(Wkv, Wkvb, 2048LL * 512);
  cvt(Wq,  Wqb,  1536LL * 2048);
  cvt(Wqr, Wqrb, 1536LL * 1024);
  cvt(Wk,  Wkb,  512LL * 2048);
  cvt(Wv,  Wvb,  512LL * 2048);
  cvt(Wkr, Wkrb, 2048LL * 1024);
  cvt(Wo,  Wob,  2048LL * 2048);

  // projections (bf16 x bf16 -> bf16, f32 accumulate inside)
  gemm_b(xb,   Wlqb, cqb,   nullptr, kRows, 1536, 2048);   // cq
  gemm_b(xb,   Wkvb, ckvb,  nullptr, kRows,  512, 2048);   // ckv
  gemm_b(cqb,  Wqb,  tmpb,  nullptr, kRows, 2048, 1536);   // q
  gemm_b(cqb,  Wqrb, tmprb, bqr,     kRows, 1024, 1536);   // qr (+bias)
  rope_concat<<<49152, blk, 0, stream>>>(tmpb, tmprb, fcos, fsin, qfb);
  gemm_b(ckvb, Wkb,  tmpb,  nullptr, kRows, 2048,  512);   // k
  gemm_b(xb,   Wkrb, tmprb, bkr,     kRows, 1024, 2048);   // kr (+bias)
  rope_concat<<<49152, blk, 0, stream>>>(tmpb, tmprb, fcos, fsin, kfb);
  gemm_b(ckvb, Wvb,  tmpb,  nullptr, kRows, 2048,  512);   // v
  v_transpose<<<32768, blk, 0, stream>>>(tmpb, vtb);

  // causal flash attention -> bf16 attn-out in tmpb
  flash_attn<<<dim3(kS / 128, kB * kH), blk, 0, stream>>>(qfb, kfb, vtb, tmpb);

  // final projection with bias -> f32 output
  {
    dim3 grid(2048 / BN, kRows / BM);
    gemm_bf16<float><<<grid, blk, 0, stream>>>(tmpb, Wob, outp, bo,
                                               kRows, 2048, 2048);
  }
}
